// GINModel_31086973288700
// MI455X (gfx1250) — matrix operand order repaired and checked by reading, not compiled, not run on hardware
//
#include <hip/hip_runtime.h>

#define N_NODES 50000
#define N_EDGES 800000
#define DIM     128
#define NLAYERS 3
#define NGRAPHS 64
#define NOUT    10

typedef __attribute__((ext_vector_type(16))) __bf16 v16bf;
typedef __attribute__((ext_vector_type(8)))  __bf16 v8bf;
typedef __attribute__((ext_vector_type(8)))  float  v8f;
typedef __attribute__((ext_vector_type(4)))  int    i4;

#if defined(__AMDGCN__) && __has_builtin(__builtin_amdgcn_global_load_async_to_lds_b128)
#define ASYNC_LDS 1
#else
#define ASYNC_LDS 0
#endif
#if defined(__AMDGCN__) && __has_builtin(__builtin_amdgcn_s_wait_asynccnt)
#define HAVE_WAIT_ASYNC 1
#else
#define HAVE_WAIT_ASYNC 0
#endif

#if ASYNC_LDS
typedef __attribute__((address_space(1))) i4* g4p;   // global int4*
typedef __attribute__((address_space(3))) i4* l4p;   // LDS int4*
#endif

// 16B global->LDS copy: async DMA path if the toolchain exposes it, else via VGPRs.
__device__ __forceinline__ void copy16_g2l(const __bf16* g, __bf16* l) {
#if ASYNC_LDS
  __builtin_amdgcn_global_load_async_to_lds_b128(
      (g4p)(uintptr_t)g, (l4p)(unsigned)(uintptr_t)l, 0, 0);
#else
  *(v8bf*)l = *(const v8bf*)g;
#endif
}

__device__ __forceinline__ void wait_async_all() {
#if ASYNC_LDS
#if HAVE_WAIT_ASYNC
  __builtin_amdgcn_s_wait_asynccnt(0);
#else
  asm volatile("s_wait_asynccnt 0" ::: "memory");
#endif
#endif
}

__device__ __forceinline__ void atomic_add_f32(float* p, float v) {
  __hip_atomic_fetch_add(p, v, __ATOMIC_RELAXED, __HIP_MEMORY_SCOPE_AGENT);
}

// ---------------- utility kernels ----------------

__global__ void k_zero_f32(float* __restrict__ p, int n) {
  int i = blockIdx.x * 256 + threadIdx.x;
  if (i < n) p[i] = 0.0f;
}

__global__ void k_f32_to_bf16(const float* __restrict__ in, __bf16* __restrict__ out, int n) {
  int i = blockIdx.x * 256 + threadIdx.x;
  if (i < n) out[i] = (__bf16)in[i];
}

// Pack row-major f32 W[K x N] into WMMA bf16 B-operand tiles.
// B operand (32x16, K x N) per-lane layout: lane holds column n = lane&15,
// k-half = lane>>4; element e (0..15) = W[kc*32 + (lane>>4)*16 + e][n].
// out[(((kc*NT)+nt)*32 + lane)*16 + e]
__global__ void k_pack_b(const float* __restrict__ W, __bf16* __restrict__ out, int K, int N) {
  int idx = blockIdx.x * 256 + threadIdx.x;
  if (idx >= K * N) return;
  int e    = idx & 15;
  int lane = (idx >> 4) & 31;
  int rest = idx >> 9;            // kc*NT + nt
  int NT   = N >> 4;
  int nt   = rest % NT;
  int kc   = rest / NT;
  int k    = kc * 32 + ((lane >> 4) << 4) + e;
  int n    = (nt << 4) + (lane & 15);
  out[idx] = (__bf16)W[k * N + n];
}

// ---------------- WMMA GEMM ----------------
// C[M x N] = A[M x K](bf16) * Bp(packed bf16) + bias ; optional += existing C.
// One wave computes a 16-row x 128-col tile (8 n-tiles). K = KC*32, M % 16 == 0.
// The 8-ntile B block is staged in LDS once per workgroup (async-to-LDS when
// available); A chunks for the whole K loop are preloaded into registers, so the
// steady state is ds_load + v_wmma only.
template <int KC>
__global__ __launch_bounds__(256)
void k_gemm_bf16(const __bf16* __restrict__ A, const __bf16* __restrict__ Bp,
                 const float* __restrict__ bias, float* __restrict__ C,
                 int M, int N, int addC) {
  extern __shared__ __bf16 Bl[];              // KC * 8 tiles * 512 elems
  const int K      = KC * 32;
  const int lane   = threadIdx.x & 31;
  const int wav    = threadIdx.x >> 5;
  const int mtile  = blockIdx.x * 8 + wav;
  const int Mtiles = M >> 4;
  const int NT     = N >> 4;
  const int ntbase = blockIdx.y * 8;

  // ---- stage the B block (all waves participate; barrier before any exit) ----
  {
    const __bf16* src = Bp + ((size_t)ntbase) * 512;   // + kc*NT*512 below
    for (int c = threadIdx.x; c < KC * 512; c += 256) {
      int kc  = c >> 9;
      int off = (c & 511) * 8;
      copy16_g2l(src + (size_t)kc * NT * 512 + off, Bl + kc * 4096 + off);
    }
    wait_async_all();
  }
  __syncthreads();
  if (mtile >= Mtiles) return;               // wave-uniform: EXEC stays all-ones

  const int half = lane >> 4;
  const int ncol = lane & 15;
  const int mrow = (mtile << 4) + ncol;

  // ---- preload all A chunks for this row-tile ----
  // A operand, 16-bit 16x32 layout (matches packed B): elems 0..15 =
  // K[kc*32 + 16*half ..] -> two contiguous 16B loads.
  v16bf areg[KC];
#pragma unroll
  for (int kc = 0; kc < KC; ++kc) {
    const __bf16* ap = A + (size_t)mrow * K + kc * 32 + half * 16;
    v8bf alo = *(const v8bf*)(ap);
    v8bf ahi = *(const v8bf*)(ap + 8);
    areg[kc] = __builtin_shufflevector(alo, ahi, 0,1,2,3,4,5,6,7,8,9,10,11,12,13,14,15);
  }

  v8f acc[8];
#pragma unroll
  for (int t = 0; t < 8; ++t) {
    float bv = bias[(ntbase + t) * 16 + ncol];
#pragma unroll
    for (int r = 0; r < 8; ++r) acc[t][r] = bv;
  }

#pragma unroll
  for (int kc = 0; kc < KC; ++kc) {
#pragma unroll
    for (int t = 0; t < 8; ++t) {
      const __bf16* bp = Bl + ((kc * 8 + t) * 32 + lane) * 16;
      v8bf blo = *(const v8bf*)(bp);
      v8bf bhi = *(const v8bf*)(bp + 8);
      v16bf b = __builtin_shufflevector(blo, bhi, 0,1,2,3,4,5,6,7,8,9,10,11,12,13,14,15);
      acc[t] = __builtin_amdgcn_wmma_f32_16x16x32_bf16(
          false, areg[kc], false, b, (short)0, acc[t], false, false);
    }
  }

  // C/D layout: vgpr r, lanes<16 -> M=r ; lanes>=16 -> M=8+r ; N = lane&15.
#pragma unroll
  for (int t = 0; t < 8; ++t) {
    int col = (ntbase + t) * 16 + ncol;
#pragma unroll
    for (int r = 0; r < 8; ++r) {
      int row = (mtile << 4) + half * 8 + r;
      size_t idx = (size_t)row * N + col;
      float v = acc[t][r];
      if (addC) v += C[idx];
      C[idx] = v;
    }
  }
}

// ---------------- graph ops ----------------

// agg[dst] += relu(h[src]) ; one wave per edge (32 lanes x float4 = 512B row).
__global__ __launch_bounds__(256)
void k_edge_agg(const float* __restrict__ h, const int* __restrict__ ei,
                float* __restrict__ agg) {
  long long gid = (long long)blockIdx.x * 256 + threadIdx.x;
  int e = (int)(gid >> 5);
  if (e >= N_EDGES) return;
  int c   = ((int)gid & 31) * 4;
  int src = ei[e];
  int dst = ei[N_EDGES + e];
  const float4 v = *(const float4*)(h + (size_t)src * DIM + c);
  float* a = agg + (size_t)dst * DIM + c;
  atomic_add_f32(a + 0, fmaxf(v.x, 0.f));
  atomic_add_f32(a + 1, fmaxf(v.y, 0.f));
  atomic_add_f32(a + 2, fmaxf(v.z, 0.f));
  atomic_add_f32(a + 3, fmaxf(v.w, 0.f));
}

__global__ void k_make_zin(const float* __restrict__ h, const float* __restrict__ agg,
                           const float* __restrict__ eps, int layer,
                           __bf16* __restrict__ out, int n) {
  int i = blockIdx.x * 256 + threadIdx.x;
  if (i >= n) return;
  float s = 1.0f + eps[layer];
  out[i] = (__bf16)(s * h[i] + agg[i]);
}

// LayerNorm(256) + relu + bf16 convert, one wave per row.
__global__ __launch_bounds__(256)
void k_ln_relu_bf16(const float* __restrict__ z, const float* __restrict__ gamma,
                    const float* __restrict__ beta, __bf16* __restrict__ out, int M) {
  int w    = (blockIdx.x * 256 + threadIdx.x) >> 5;
  int lane = threadIdx.x & 31;
  if (w >= M) return;
  const float* row = z + (size_t)w * 256;
  float v[8];
  float s = 0.f, s2 = 0.f;
#pragma unroll
  for (int j = 0; j < 8; ++j) {
    float x = row[lane * 8 + j];
    v[j] = x; s += x; s2 += x * x;
  }
#pragma unroll
  for (int off = 16; off > 0; off >>= 1) {
    s  += __shfl_xor(s,  off, 32);
    s2 += __shfl_xor(s2, off, 32);
  }
  float mu  = s  * (1.0f / 256.0f);
  float var = s2 * (1.0f / 256.0f) - mu * mu;
  float rs  = rsqrtf(var + 1e-5f);
#pragma unroll
  for (int j = 0; j < 8; ++j) {
    int col = lane * 8 + j;
    float y = (v[j] - mu) * rs * gamma[col] + beta[col];
    out[(size_t)w * 256 + col] = (__bf16)fmaxf(y, 0.f);
  }
}

__global__ void k_pool(const float* __restrict__ h, const int* __restrict__ batch,
                       float* __restrict__ g) {
  int i = blockIdx.x * 256 + threadIdx.x;
  if (i >= N_NODES * DIM) return;
  int node = i >> 7;
  int c    = i & 127;
  atomic_add_f32(&g[(size_t)batch[node] * DIM + c], h[i]);
}

__global__ void k_out1(const float* __restrict__ g, const float* __restrict__ w1,
                       const float* __restrict__ b1, float* __restrict__ t) {
  int i = blockIdx.x * 256 + threadIdx.x;   // 64*256
  if (i >= NGRAPHS * 256) return;
  int gi = i >> 8, j = i & 255;
  float acc = b1[j];
  for (int k = 0; k < DIM; ++k) acc += g[gi * DIM + k] * w1[k * 256 + j];
  t[i] = fmaxf(acc, 0.f);
}

__global__ void k_out2(const float* __restrict__ t, const float* __restrict__ w2,
                       const float* __restrict__ b2, float* __restrict__ out) {
  int i = blockIdx.x * 64 + threadIdx.x;    // 640
  if (i >= NGRAPHS * NOUT) return;
  int gi = i / NOUT, j = i % NOUT;
  float acc = b2[j];
  for (int k = 0; k < 256; ++k) acc += t[gi * 256 + k] * w2[k * NOUT + j];
  out[i] = acc;
}

// ---------------- launcher ----------------

extern "C" void kernel_launch(void* const* d_in, const int* in_sizes, int n_in,
                              void* d_out, int out_size, void* d_ws, size_t ws_size,
                              hipStream_t stream) {
  const float* x        = (const float*)d_in[0];
  const int*   ei       = (const int*)  d_in[1];
  const int*   batch    = (const int*)  d_in[2];
  const float* lin_w    = (const float*)d_in[3];
  const float* lin_b    = (const float*)d_in[4];
  const float* conv_w1  = (const float*)d_in[5];
  const float* conv_b1  = (const float*)d_in[6];
  const float* ln_g     = (const float*)d_in[7];
  const float* ln_b     = (const float*)d_in[8];
  const float* conv_w2  = (const float*)d_in[9];
  const float* conv_b2  = (const float*)d_in[10];
  const float* conv_eps = (const float*)d_in[11];
  const float* out_w1   = (const float*)d_in[12];
  const float* out_b1   = (const float*)d_in[13];
  const float* out_w2   = (const float*)d_in[14];
  const float* out_b2   = (const float*)d_in[15];
  float* out = (float*)d_out;

  char* ws = (char*)d_ws;
  float*  h     = (float*) (ws + 0);          // 25.6 MB
  float*  agg   = (float*) (ws + 25600000);   // 25.6 MB
  float*  z1    = (float*) (ws + 51200000);   // 51.2 MB
  __bf16* ab    = (__bf16*)(ws + 102400000);  // 25.6 MB (bf16 A-operand staging)
  __bf16* linp  = (__bf16*)(ws + 128000000);  // 32 KB
  __bf16* w1p   = (__bf16*)(ws + 128032768);  // 3 * 64 KB
  __bf16* w2p   = (__bf16*)(ws + 128229376);  // 3 * 64 KB
  float*  gpool = (float*) (ws + 128425984);  // 32 KB
  float*  thid  = (float*) (ws + 128458752);  // 64 KB

  const int nhd = N_NODES * DIM;              // 6.4M

  // Prepack weights into WMMA B-operand layout (tiny, once per launch).
  k_pack_b<<<(128 * 128 + 255) / 256, 256, 0, stream>>>(lin_w, linp, 128, 128);
  for (int l = 0; l < NLAYERS; ++l) {
    k_pack_b<<<(128 * 256 + 255) / 256, 256, 0, stream>>>(conv_w1 + l * 128 * 256,
                                                          w1p + l * 128 * 256, 128, 256);
    k_pack_b<<<(256 * 128 + 255) / 256, 256, 0, stream>>>(conv_w2 + l * 256 * 128,
                                                          w2p + l * 256 * 128, 256, 128);
  }

  // h = x @ lin_in_w + b  (bf16 WMMA, f32 accumulate)
  k_f32_to_bf16<<<(nhd + 255) / 256, 256, 0, stream>>>(x, ab, nhd);
  k_gemm_bf16<4><<<dim3(391, 1), 256, 32768, stream>>>(ab, linp, lin_b, h,
                                                       N_NODES, 128, 0);

  for (int l = 0; l < NLAYERS; ++l) {
    k_zero_f32<<<(nhd + 255) / 256, 256, 0, stream>>>(agg, nhd);
    k_edge_agg<<<(N_EDGES * 32) / 256, 256, 0, stream>>>(h, ei, agg);
    k_make_zin<<<(nhd + 255) / 256, 256, 0, stream>>>(h, agg, conv_eps, l, ab, nhd);
    k_gemm_bf16<4><<<dim3(391, 2), 256, 32768, stream>>>(ab, w1p + l * 128 * 256,
                                                         conv_b1 + l * 256, z1,
                                                         N_NODES, 256, 0);
    k_ln_relu_bf16<<<(N_NODES + 7) / 8, 256, 0, stream>>>(z1, ln_g + l * 256,
                                                          ln_b + l * 256, ab, N_NODES);
    k_gemm_bf16<8><<<dim3(391, 1), 256, 65536, stream>>>(ab, w2p + l * 256 * 128,
                                                         conv_b2 + l * 128, h,
                                                         N_NODES, 128, 1);  // h += z
  }

  k_zero_f32<<<(NGRAPHS * DIM + 255) / 256, 256, 0, stream>>>(gpool, NGRAPHS * DIM);
  k_pool<<<(nhd + 255) / 256, 256, 0, stream>>>(h, batch, gpool);
  k_out1<<<(NGRAPHS * 256 + 255) / 256, 256, 0, stream>>>(gpool, out_w1, out_b1, thid);
  k_out2<<<(NGRAPHS * NOUT + 63) / 64, 64, 0, stream>>>(thid, out_w2, out_b2, out);
}
